// PointNet2Encoder_31215822307857
// MI455X (gfx1250) — matrix operand order repaired and checked by reading, not compiled, not run on hardware
//
#include <hip/hip_runtime.h>
#include <hip/hip_bf16.h>

// ---------------------------------------------------------------------------
// PointNet++ encoder for MI455X (gfx1250): wave-per-point WMMA bf16 pipeline.
// Layer-1 computed operand-swapped (D' = W^T x feats^T) so its C-tiles map
// directly into layer-2 A-fragments -> h stays register-resident (no LDS
// round trip). Layer-2 streams W4^T through LDS in two 128KB column blocks.
// ---------------------------------------------------------------------------

typedef __attribute__((ext_vector_type(16))) __bf16 v16bf;
typedef __attribute__((ext_vector_type(8)))  float  v8f;

#define P_TOTAL 32768
#define KNBR    16

// ---- workspace layout (bytes) ----
#define OFF_X1   0ull            // [P][128] bf16            : 8,388,608
#define OFF_W1T  8388608ull      // [64][Kpad=32]  bf16      : 4,096
#define OFF_W2T  8392704ull      // [128][Kpad=64] bf16      : 16,384
#define OFF_W3T  8409088ull      // [256][Kpad=160] bf16     : 81,920
#define OFF_W4T  8491008ull      // [512][Kpad=256] bf16     : 262,144

// ---- dynamic LDS layout for sa2 ----
#define SM_W3T 0                  // 81,920 B (resident)
#define SM_W4B 81920              // 131,072 B (one 256-col block of W4^T)
#define SA2_LDS 212992

static __device__ __forceinline__ v8f wmma_bf16(v16bf a, v16bf b, v8f c) {
  // D = A(16x32 bf16) * B(32x16 bf16) + C(16x16 f32)
  return __builtin_amdgcn_wmma_f32_16x16x32_bf16(false, a, false, b,
                                                 (short)0, c, false, false);
}

// one contiguous 32-byte fragment read (two b128)
static __device__ __forceinline__ v16bf ld_frag32(const char* p) {
  return *(const v16bf*)p;
}
// fragment from two 16-byte chunks
static __device__ __forceinline__ v16bf ld_frag16x2(const char* p0, const char* p1) {
  union { v16bf v; uint4 q[2]; } u;
  u.q[0] = *(const uint4*)p0;
  u.q[1] = *(const uint4*)p1;
  return u.v;
}
static __device__ __forceinline__ v16bf zero_frag() {
  union { v16bf v; uint4 q[2]; } u;
  u.q[0] = uint4{0u, 0u, 0u, 0u};
  u.q[1] = uint4{0u, 0u, 0u, 0u};
  return u.v;
}
// B'-layout tile of feats^T holding (dx,dy,dz) in K slots 0..2 (low lanes)
static __device__ __forceinline__ v16bf diff_frag(float dx, float dy, float dz, bool act) {
  v16bf a = zero_frag();
  if (act) { a[0] = (__bf16)dx; a[1] = (__bf16)dy; a[2] = (__bf16)dz; }
  return a;
}
// A-layout fragment of a W^T[N][Kpad] LDS image: c = absolute column,
// lane M = lane&15, K-halves selected by lane>=16
static __device__ __forceinline__ v16bf ld_afragT(const char* base, int c, int kt,
                                                  int rowBytes, bool hi) {
  const char* p0 = base + c * rowBytes + kt * 64 + (hi ? 16 : 0);
  return ld_frag16x2(p0, p0 + 32);
}

// float -> order-preserving uint (for atomic max)
static __device__ __forceinline__ unsigned f2mono(float f) {
  unsigned b = __float_as_uint(f);
  return (b & 0x80000000u) ? ~b : (b | 0x80000000u);
}

// ---------------------------------------------------------------------------
// Weight prep: transpose to bf16 W^T[N][Kpad], zero-padded K
// ---------------------------------------------------------------------------
__global__ void prep_weights(const float* __restrict__ W1, const float* __restrict__ W2,
                             const float* __restrict__ W3, const float* __restrict__ W4,
                             char* __restrict__ ws) {
  int i = blockIdx.x * 256 + threadIdx.x;
  if (i < 64 * 32) {                       // W1 [3][64] -> W1T [64][32]
    int c = i >> 5, k = i & 31;
    float v = (k < 3) ? W1[k * 64 + c] : 0.f;
    *(__bf16*)(ws + OFF_W1T + (size_t)i * 2) = (__bf16)v;
  }
  int j = i - 64 * 32;
  if (j >= 0 && j < 128 * 64) {            // W2 [64][128] -> W2T [128][64]
    int c = j >> 6, k = j & 63;
    *(__bf16*)(ws + OFF_W2T + (size_t)j * 2) = (__bf16)W2[k * 128 + c];
  }
  int m = i - 64 * 32 - 128 * 64;
  if (m >= 0 && m < 256 * 160) {           // W3 [131][256] -> W3T [256][160]
    int c = m / 160, k = m % 160;
    float v = (k < 131) ? W3[k * 256 + c] : 0.f;
    *(__bf16*)(ws + OFF_W3T + (size_t)m * 2) = (__bf16)v;
  }
  int n = i - 64 * 32 - 128 * 64 - 256 * 160;
  if (n >= 0 && n < 512 * 256) {           // W4 [256][512] -> W4T [512][256]
    int c = n >> 8, k = n & 255;
    *(__bf16*)(ws + OFF_W4T + (size_t)n * 2) = (__bf16)W4[k * 512 + c];
  }
}

__global__ void init_out(unsigned* __restrict__ outU) {
  int i = blockIdx.x * 256 + threadIdx.x;
  if (i < 16 * 512) outU[i] = 0u;          // monotone -inf sentinel
}

__global__ void final_out(float* __restrict__ outF) {
  int i = blockIdx.x * 256 + threadIdx.x;
  if (i < 16 * 512) {
    unsigned u = ((const unsigned*)outF)[i];
    unsigned b = (u & 0x80000000u) ? (u & 0x7fffffffu) : ~u;
    outF[i] = __uint_as_float(b);
  }
}

// ---------------------------------------------------------------------------
// sa1: x1[p] = max_k relu( (pos_j - pos_i) W1 + b1 ) W2 + b2     -> ws bf16
// layer1 swapped: D' = W1T-tiles (A) x diff^T (B); C-tiles pack straight
// into layer-2 A-fragments. No LDS h scratch at all.
// ---------------------------------------------------------------------------
__global__ void __launch_bounds__(256) sa1_kernel(const float* __restrict__ pos,
                                                  const long long* __restrict__ nbr,
                                                  const float* __restrict__ b1,
                                                  const float* __restrict__ b2,
                                                  char* __restrict__ ws) {
  __shared__ char sm[4096 + 16384];            // W1T | W2T
  const int tid = threadIdx.x;
  {  // coop stage W1T+W2T (contiguous in ws)
    const uint4* src = (const uint4*)(ws + OFF_W1T);
    uint4* dst = (uint4*)sm;
    for (int i = tid; i < (4096 + 16384) / 16; i += 256) dst[i] = src[i];
  }
  __syncthreads();
  const char* w1t = sm;
  const char* w2t = sm + 4096;
  const int  lane = tid & 31;
  const int  row  = lane & 15;
  const bool hi   = lane >= 16;
  char* x1 = ws + OFF_X1;

  for (int g = 0; g < 8; ++g) {
    int p = blockIdx.x * 64 + g * 8 + (tid >> 5);
    int nidx = (int)nbr[(size_t)p * KNBR + row];
    float cx = pos[3 * p], cy = pos[3 * p + 1], cz = pos[3 * p + 2];
    // B'-fragment: feats^T, lane N = neighbor row, K=0..2 valid on low lanes
    v16bf bgd = diff_frag(pos[3 * nidx] - cx, pos[3 * nidx + 1] - cy,
                          pos[3 * nidx + 2] - cz, !hi);
    // ---- layer 1 (swapped): 4 col-tiles -> 2 layer-2 A-fragments
    v16bf a2f[2];
#pragma unroll
    for (int t = 0; t < 2; ++t) {
      v8f e = {}, o = {};
      e = wmma_bf16(ld_afragT(w1t, (2 * t) * 16 + row, 0, 64, hi), bgd, e);
      o = wmma_bf16(ld_afragT(w1t, (2 * t + 1) * 16 + row, 0, 64, hi), bgd, o);
      const float* be = b1 + 32 * t + (hi ? 8 : 0);   // col = 32t + 8hi + r
      v16bf a;
#pragma unroll
      for (int r = 0; r < 8; ++r) {
        a[r]     = (__bf16)fmaxf(e[r] + be[r], 0.f);
        a[r + 8] = (__bf16)fmaxf(o[r] + be[r + 16], 0.f);
      }
      a2f[t] = a;
    }
    // ---- layer 2 (unswapped): [16x64]@[64x128], row-max -> x1
#pragma unroll
    for (int jt = 0; jt < 8; ++jt) {
      int col = jt * 16 + row;
      v8f acc = {};
      acc = wmma_bf16(a2f[0], ld_afragT(w2t, col, 0, 128, hi), acc);
      acc = wmma_bf16(a2f[1], ld_afragT(w2t, col, 1, 128, hi), acc);
      float m = acc[0];
#pragma unroll
      for (int r = 1; r < 8; ++r) m = fmaxf(m, acc[r]);
      m = fmaxf(m, __shfl_xor(m, 16, 32));   // merge M=0..7 / M=8..15 halves
      m += b2[col];
      if (!hi) *(__bf16*)(x1 + (size_t)p * 256 + col * 2) = (__bf16)m;
    }
  }
}

// ---------------------------------------------------------------------------
// sa2: x2[p] = max_k relu([x1_j, diff] W3 + b3) W4 + b4; segment-max into out.
// layer1 swapped (h register-resident); W3T resident in LDS; W4T streamed in
// two 128KB column blocks through LDS.
// ---------------------------------------------------------------------------
__global__ void __launch_bounds__(256) sa2_kernel(const float* __restrict__ pos,
                                                  const long long* __restrict__ nbr,
                                                  const float* __restrict__ b3,
                                                  const float* __restrict__ b4,
                                                  const char* __restrict__ ws,
                                                  unsigned* __restrict__ outU) {
  extern __shared__ char sm[];
  const int tid = threadIdx.x;
  {  // coop stage W3T (80 KB), resident all kernel
    const uint4* src = (const uint4*)(ws + OFF_W3T);
    uint4* dst = (uint4*)(sm + SM_W3T);
    for (int i = tid; i < 81920 / 16; i += 256) dst[i] = src[i];
  }
  const int  lane = tid & 31;
  const int  wv   = tid >> 5;
  const int  row  = lane & 15;
  const bool hi   = lane >= 16;
  const char* x1 = ws + OFF_X1;
  __syncthreads();

  for (int g = 0; g < 8; ++g) {
    int p = blockIdx.x * 64 + g * 8 + wv;
    int nidx = (int)nbr[(size_t)p * KNBR + row];
    // ---- B'-fragments of feats^T (Kpad=160): lane N = neighbor row,
    //      K in table order -> two 16B gathers per K-tile from x1[nbr]
    const char* xr = x1 + (size_t)nidx * 256;
    v16bf bg[5];
#pragma unroll
    for (int t = 0; t < 4; ++t) {
      const char* pb = xr + t * 64 + (hi ? 16 : 0);
      bg[t] = ld_frag16x2(pb, pb + 32);
    }
    {
      float cx = pos[3 * p], cy = pos[3 * p + 1], cz = pos[3 * p + 2];
      bg[4] = diff_frag(pos[3 * nidx] - cx, pos[3 * nidx + 1] - cy,
                        pos[3 * nidx + 2] - cz, !hi);
    }
    // ---- layer 1 (swapped): 16 col-tiles -> 8 layer-2 A-fragments, in regs
    v16bf a2[8];
#pragma unroll
    for (int t = 0; t < 8; ++t) {
      v8f e = {}, o = {};
#pragma unroll
      for (int kt = 0; kt < 5; ++kt) {
        e = wmma_bf16(ld_afragT(sm + SM_W3T, (2 * t) * 16 + row, kt, 320, hi),
                      bg[kt], e);
        o = wmma_bf16(ld_afragT(sm + SM_W3T, (2 * t + 1) * 16 + row, kt, 320, hi),
                      bg[kt], o);
      }
      const float* be = b3 + 32 * t + (hi ? 8 : 0);   // col = 32t + 8hi + r
      v16bf a;
#pragma unroll
      for (int r = 0; r < 8; ++r) {
        a[r]     = (__bf16)fmaxf(e[r] + be[r], 0.f);
        a[r + 8] = (__bf16)fmaxf(o[r] + be[r + 16], 0.f);
      }
      a2[t] = a;
    }

    // ---- layer 2 (unswapped): [16x256]@[256x512] in two 256-col blocks
    int b = p >> 11;  // 2048 points per cloud
    for (int nblk = 0; nblk < 2; ++nblk) {
      __syncthreads();                       // protect previous block's readers
      {                                      // coop stage W4T cols [nblk*256, +256)
        const uint4* src = (const uint4*)(ws + OFF_W4T + (size_t)nblk * 131072);
        uint4* dst = (uint4*)(sm + SM_W4B);
        for (int i = tid; i < 131072 / 16; i += 256) dst[i] = src[i];
      }
      __syncthreads();
#pragma unroll 2
      for (int j2 = 0; j2 < 16; ++j2) {
        int cl = j2 * 16 + row;
        v8f acc = {};
#pragma unroll
        for (int t = 0; t < 8; ++t)
          acc = wmma_bf16(a2[t],
                          ld_afragT(sm + SM_W4B, cl, t, 512, hi),
                          acc);
        float m = acc[0];
#pragma unroll
        for (int r = 1; r < 8; ++r) m = fmaxf(m, acc[r]);
        m = fmaxf(m, __shfl_xor(m, 16, 32));
        int col = nblk * 256 + cl;
        m += b4[col];
        if (!hi) atomicMax(outU + b * 512 + col, f2mono(m));  // segment max
      }
    }
  }
}

// ---------------------------------------------------------------------------
extern "C" void kernel_launch(void* const* d_in, const int* in_sizes, int n_in,
                              void* d_out, int out_size, void* d_ws, size_t ws_size,
                              hipStream_t stream) {
  (void)in_sizes; (void)n_in; (void)out_size; (void)ws_size;
  const float*     pos = (const float*)d_in[0];
  const long long* nbr = (const long long*)d_in[1];
  // d_in[2] = batch vector (implicit: b = p >> 11)
  const float* W1 = (const float*)d_in[3];
  const float* b1 = (const float*)d_in[4];
  const float* W2 = (const float*)d_in[5];
  const float* b2 = (const float*)d_in[6];
  const float* W3 = (const float*)d_in[7];
  const float* b3 = (const float*)d_in[8];
  const float* W4 = (const float*)d_in[9];
  const float* b4 = (const float*)d_in[10];
  char* ws = (char*)d_ws;

  hipFuncSetAttribute((const void*)sa2_kernel,
                      hipFuncAttributeMaxDynamicSharedMemorySize, SA2_LDS);

  prep_weights<<<712, 256, 0, stream>>>(W1, W2, W3, W4, ws);
  init_out<<<32, 256, 0, stream>>>((unsigned*)d_out);
  sa1_kernel<<<512, 256, 0, stream>>>(pos, nbr, b1, b2, ws);
  sa2_kernel<<<512, 256, SA2_LDS, stream>>>(pos, nbr, b3, b4, ws, (unsigned*)d_out);
  final_out<<<32, 256, 0, stream>>>((float*)d_out);
}
